// MultiHeadAttention_91061896610451
// MI455X (gfx1250) — matrix operand order repaired and checked by reading, not compiled, not run on hardware
//
#include <hip/hip_runtime.h>

// ---------------------------------------------------------------------------
// MHA forward for MI455X (gfx1250, wave32, WMMA 16x16x32 f16 -> f32 acc)
// Async global->LDS staging (ASYNCcnt) + double-buffered LDS pipelines.
// ---------------------------------------------------------------------------

typedef __attribute__((ext_vector_type(16))) _Float16 v16h;
typedef __attribute__((ext_vector_type(8)))  _Float16 v8h;
typedef __attribute__((ext_vector_type(8)))  float    v8f;

static __device__ __forceinline__ v16h cat8(v8h lo, v8h hi) {
    return __builtin_shufflevector(lo, hi, 0,1,2,3,4,5,6,7,8,9,10,11,12,13,14,15);
}

static __device__ __forceinline__ v8f wmma_f16(v16h a, v16h b, v8f c) {
    return __builtin_amdgcn_wmma_f32_16x16x32_f16(false, a, false, b, (short)0, c,
                                                  false, false);
}

// 32-byte per-lane async copy: global -> LDS (2x b128, tracked by ASYNCcnt).
// Low 32 bits of a generic shared pointer are the wave-relative LDS address.
static __device__ __forceinline__ void async_copy32B(const _Float16* g, _Float16* l) {
    const unsigned off = (unsigned)(uintptr_t)l;
    asm volatile(
        "global_load_async_to_lds_b128 %0, %1, off\n\t"
        "global_load_async_to_lds_b128 %2, %3, off"
        :
        : "v"(off), "v"(g), "v"(off + 16u), "v"(g + 8)
        : "memory");
}
static __device__ __forceinline__ void wait_async0() {
    asm volatile("s_wait_asynccnt 0" ::: "memory");
}

// Problem constants
#define D_MODEL 1024
#define NHEAD   16
#define DK      64
#define SEQ     2048
#define BATCH   4

// ---------------------------------------------------------------------------
// f32 -> f16 elementwise conversion (8 elems/thread, vector store)
// ---------------------------------------------------------------------------
__global__ __launch_bounds__(256) void cvt_kernel(const float* __restrict__ src,
                                                  _Float16* __restrict__ dst, int n8)
{
    const int i = blockIdx.x * blockDim.x + threadIdx.x;
    if (i < n8) {
        const float* s = src + (size_t)i * 8;
        v8h d;
        #pragma unroll
        for (int j = 0; j < 8; ++j) d[j] = (_Float16)s[j];
        *(v8h*)(dst + (size_t)i * 8) = d;
    }
}

// ---------------------------------------------------------------------------
// GEMM: C[M,N] = A[M,K] * W[N,K]^T + bias[N]   (A, W already f16)
//   HEADS_OUT: store f16 into [B,H,S,DK] head layout, else f32 row-major [M,N]
// Block: 256 threads = 8 waves (4 M x 2 N), tile 128x128, per-wave 32x64,
// K-step 32 -> 8 WMMA per wave per k-step. Double-buffered async staging.
// ---------------------------------------------------------------------------
#define BM 128
#define BN 128
#define BK 32
#define AST (BK + 8)   // 40 f16 = 80 B rows (multiple of 16 B)
#define BST (BK + 8)

template <bool HEADS_OUT>
__global__ __launch_bounds__(256) void gemm_kernel(
    const _Float16* __restrict__ A, const _Float16* __restrict__ W,
    const float* __restrict__ bias, void* __restrict__ Cvp,
    int M, int N, int K)
{
    __shared__ _Float16 As[2][BM * AST];
    __shared__ _Float16 Bs[2][BN * BST];

    const int tid   = threadIdx.x;
    const int lane  = tid & 31;
    const int wid   = tid >> 5;
    const int waveM = wid & 3;      // 0..3  -> 32 rows each
    const int waveN = wid >> 2;     // 0..1  -> 64 cols each
    const int hi    = lane >> 4;    // half-wave select
    const int lo16  = lane & 15;

    const int m0 = blockIdx.x * BM;
    const int n0 = blockIdx.y * BN;

    // staging coords: 2 threads/row, 32 B each
    const int srow = tid >> 1;
    const int sc0  = (tid & 1) * 16;
    const _Float16* Ag = &A[(size_t)(m0 + srow) * K + sc0];
    const _Float16* Wg = &W[(size_t)(n0 + srow) * K + sc0];
    _Float16* Asl0 = &As[0][srow * AST + sc0];
    _Float16* Bsl0 = &Bs[0][srow * BST + sc0];
    _Float16* Asl1 = &As[1][srow * AST + sc0];
    _Float16* Bsl1 = &Bs[1][srow * BST + sc0];

    v8f acc[2][4] = {};

    // prologue: stage k0 = 0 into buffer 0
    async_copy32B(Ag, Asl0);
    async_copy32B(Wg, Bsl0);

    int p = 0;
    for (int k0 = 0; k0 < K; k0 += BK) {
        wait_async0();
        __syncthreads();                       // buffer p ready for everyone

        if (k0 + BK < K) {                     // prefetch k0+BK into buffer 1-p
            async_copy32B(Ag + k0 + BK, p ? Asl0 : Asl1);
            async_copy32B(Wg + k0 + BK, p ? Bsl0 : Bsl1);
        }

        // preload the 4 B fragments (lane holds col n=lo16, k-run [16*hi,+16))
        v16h bf[4];
        #pragma unroll
        for (int tj = 0; tj < 4; ++tj) {
            const _Float16* br = &Bs[p][(waveN * 64 + tj * 16 + lo16) * BST + 16 * hi];
            bf[tj] = cat8(*(const v8h*)br, *(const v8h*)(br + 8));
        }
        #pragma unroll
        for (int ti = 0; ti < 2; ++ti) {
            // A fragment: lane holds row m=lo16, k-run [16*hi,+16)
            const _Float16* ar = &As[p][(waveM * 32 + ti * 16 + lo16) * AST + 16 * hi];
            v16h afrag = cat8(*(const v8h*)ar, *(const v8h*)(ar + 8));
            #pragma unroll
            for (int tj = 0; tj < 4; ++tj)
                acc[ti][tj] = wmma_f16(afrag, bf[tj], acc[ti][tj]);
        }
        __syncthreads();                       // done reading buffer p
        p ^= 1;
    }

    // ---- epilogue: C layout is lane=col(n=lo16), reg r = row (+8 for hi) ----
    #pragma unroll
    for (int ti = 0; ti < 2; ++ti) {
        #pragma unroll
        for (int tj = 0; tj < 4; ++tj) {
            const int ng = n0 + waveN * 64 + tj * 16 + lo16;
            const float bv = bias[ng];
            #pragma unroll
            for (int r = 0; r < 8; ++r) {
                const int mg = m0 + waveM * 32 + ti * 16 + r + 8 * hi;
                const float val = acc[ti][tj][r] + bv;
                if constexpr (HEADS_OUT) {
                    const int b = mg >> 11, s = mg & (SEQ - 1);
                    const int h = ng >> 6,  d = ng & (DK - 1);
                    ((_Float16*)Cvp)[((((size_t)b * NHEAD + h) * SEQ) + s) * DK + d] =
                        (_Float16)val;
                } else {
                    ((float*)Cvp)[(size_t)mg * N + ng] = val;
                }
            }
        }
    }
}

// ---------------------------------------------------------------------------
// Flash attention per (b, h, 128-query block).  8 waves, 16 query rows each.
// 64 keys/iter: 8 WMMA scores + 2 WMMA row-sum (P*ones) + 8 WMMA P*V.
// K staged async global->LDS, V prefetched to VGPRs; both double-buffered.
// ---------------------------------------------------------------------------
#define TKEYS 64
#define KST (DK + 8)      // 72 f16 = 144 B rows (multiple of 16 B)
#define VST (TKEYS + 8)   // 72
#define PST (TKEYS + 8)   // 72

__global__ __launch_bounds__(256) void attn_kernel(
    const _Float16* __restrict__ Qp, const _Float16* __restrict__ Kp,
    const _Float16* __restrict__ Vp, _Float16* __restrict__ ctx)
{
    __shared__ _Float16 Ks[2][TKEYS * KST];     // [key][d]
    __shared__ _Float16 Vs[2][DK * VST];        // [d][key]  (V transposed)
    __shared__ _Float16 Ps[8 * 16 * PST];       // per-wave P scratch

    const int tid  = threadIdx.x;
    const int lane = tid & 31;
    const int wid  = tid >> 5;
    const int hi   = lane >> 4;
    const int lo16 = lane & 15;

    const int h = blockIdx.y;
    const int b = blockIdx.z;
    const int q0 = blockIdx.x * 128 + wid * 16;   // this wave's 16 query rows

    const size_t headoff = ((size_t)b * NHEAD + h) * SEQ * DK;
    const _Float16* Qh = Qp + headoff;
    const _Float16* Kh = Kp + headoff;
    const _Float16* Vh = Vp + headoff;

    // Q fragments (registers), pre-scaled by 1/sqrt(DK)=0.125 (exact in f16)
    v16h qf[2];
    {
        const _Float16* qrow = Qh + (size_t)(q0 + lo16) * DK;
        #pragma unroll
        for (int kc = 0; kc < 2; ++kc) {
            v16h f = cat8(*(const v8h*)(qrow + kc * 32 + 16 * hi),
                          *(const v8h*)(qrow + kc * 32 + 16 * hi + 8));
            #pragma unroll
            for (int j = 0; j < 16; ++j) f[j] = f[j] * (_Float16)0.125f;
            qf[kc] = f;
        }
    }

    v16h onesf;
    #pragma unroll
    for (int j = 0; j < 16; ++j) onesf[j] = (_Float16)1.0f;

    float mrow[8], lrow[8];
    v8f oacc[4] = {};
    #pragma unroll
    for (int r = 0; r < 8; ++r) { mrow[r] = -1e30f; lrow[r] = 0.0f; }

    // staging coords: 4 threads/key row, 32 B each
    const int skey = tid >> 2;
    const int sc0  = (tid & 3) * 16;
    const _Float16* Kg = Kh + (size_t)skey * DK + sc0;
    const _Float16* Vg = Vh + (size_t)skey * DK + sc0;
    _Float16* Ksl[2] = { &Ks[0][skey * KST + sc0], &Ks[1][skey * KST + sc0] };

    // prologue: stage t0 = 0
    async_copy32B(Kg, Ksl[0]);
    v8h va = *(const v8h*)(Vg);
    v8h vb = *(const v8h*)(Vg + 8);

    int p = 0;
    for (int t0 = 0; t0 < SEQ; t0 += TKEYS) {
        // transpose-store this tile's V registers into Vs[p]
        #pragma unroll
        for (int i = 0; i < 8; ++i) {
            Vs[p][(sc0 + i) * VST + skey]     = va[i];
            Vs[p][(sc0 + 8 + i) * VST + skey] = vb[i];
        }
        wait_async0();
        __syncthreads();                        // Ks[p], Vs[p] ready

        if (t0 + TKEYS < SEQ) {                 // prefetch next tile
            const size_t goff = (size_t)(t0 + TKEYS) * DK;
            async_copy32B(Kg + goff, Ksl[1 - p]);
            va = *(const v8h*)(Vg + goff);
            vb = *(const v8h*)(Vg + goff + 8);
        }

        // ---- scores: S[16 x 64] = Qtile(16x64) * K^T ----
        v8f sacc[4] = {};
        #pragma unroll
        for (int tj = 0; tj < 4; ++tj) {
            #pragma unroll
            for (int kc = 0; kc < 2; ++kc) {
                const _Float16* br = &Ks[p][(tj * 16 + lo16) * KST + kc * 32 + 16 * hi];
                v16h bfr = cat8(*(const v8h*)br, *(const v8h*)(br + 8));
                sacc[tj] = wmma_f16(qf[kc], bfr, sacc[tj]);
            }
        }

        // ---- online softmax: row max via 16-lane shuffle reduce ----
        float alpha[8];
        #pragma unroll
        for (int r = 0; r < 8; ++r) {
            float mx = fmaxf(fmaxf(sacc[0][r], sacc[1][r]),
                             fmaxf(sacc[2][r], sacc[3][r]));
            #pragma unroll
            for (int off = 8; off >= 1; off >>= 1)
                mx = fmaxf(mx, __shfl_xor(mx, off, 32));
            const float mnew = fmaxf(mrow[r], mx);
            alpha[r] = __expf(mrow[r] - mnew);
            mrow[r] = mnew;
            #pragma unroll
            for (int tj = 0; tj < 4; ++tj)
                sacc[tj][r] = __expf(sacc[tj][r] - mnew);
        }

        // ---- P: C-layout regs -> LDS row-major -> A-layout fragments ----
        _Float16* P = &Ps[wid * 16 * PST];
        #pragma unroll
        for (int tj = 0; tj < 4; ++tj)
            #pragma unroll
            for (int r = 0; r < 8; ++r)
                P[(r + 8 * hi) * PST + tj * 16 + lo16] = (_Float16)sacc[tj][r];

        #pragma unroll
        for (int dn = 0; dn < 4; ++dn)
            #pragma unroll
            for (int r = 0; r < 8; ++r)
                oacc[dn][r] *= alpha[r];

        v16h pf[2];
        #pragma unroll
        for (int kc = 0; kc < 2; ++kc) {
            const _Float16* prow = &P[lo16 * PST + kc * 32 + 16 * hi];
            pf[kc] = cat8(*(const v8h*)prow, *(const v8h*)(prow + 8));
        }

        // ---- row-sum of P on the matrix pipe: rs = P * ones(64x16) ----
        v8f rsacc = {};
        rsacc = wmma_f16(pf[0], onesf, rsacc);
        rsacc = wmma_f16(pf[1], onesf, rsacc);
        #pragma unroll
        for (int r = 0; r < 8; ++r)
            lrow[r] = lrow[r] * alpha[r] + rsacc[r];

        // ---- O += P(16x64) * V(64x64), V^T staged as B(k=key, n=d) ----
        #pragma unroll
        for (int dn = 0; dn < 4; ++dn) {
            #pragma unroll
            for (int kc = 0; kc < 2; ++kc) {
                const _Float16* vr = &Vs[p][(dn * 16 + lo16) * VST + kc * 32 + 16 * hi];
                v16h vf = cat8(*(const v8h*)vr, *(const v8h*)(vr + 8));
                oacc[dn] = wmma_f16(pf[kc], vf, oacc[dn]);
            }
        }
        __syncthreads();                        // done reading buffer p
        p ^= 1;
    }

    // ---- finalize: ctx[b][s][h*64+d] f16, merged-head layout ----
    #pragma unroll
    for (int dn = 0; dn < 4; ++dn) {
        #pragma unroll
        for (int r = 0; r < 8; ++r) {
            const int s = q0 + r + 8 * hi;
            const int d = dn * 16 + lo16;
            const float v = oacc[dn][r] / lrow[r];
            ctx[((size_t)b * SEQ + s) * D_MODEL + h * DK + d] = (_Float16)v;
        }
    }
}

// ---------------------------------------------------------------------------
// Host-side launch
// inputs: 0:q 1:k 2:v 3:mask 4:Wq 5:bq 6:Wk 7:bk 8:Wv 9:bv 10:W0 11:b0
// ---------------------------------------------------------------------------
extern "C" void kernel_launch(void* const* d_in, const int* in_sizes, int n_in,
                              void* d_out, int out_size, void* d_ws, size_t ws_size,
                              hipStream_t stream)
{
    (void)in_sizes; (void)n_in; (void)out_size; (void)ws_size;

    const float* q  = (const float*)d_in[0];
    const float* k  = (const float*)d_in[1];
    const float* v  = (const float*)d_in[2];
    const float* Wq = (const float*)d_in[4];
    const float* bq = (const float*)d_in[5];
    const float* Wk = (const float*)d_in[6];
    const float* bk = (const float*)d_in[7];
    const float* Wv = (const float*)d_in[8];
    const float* bv = (const float*)d_in[9];
    const float* W0 = (const float*)d_in[10];
    const float* b0 = (const float*)d_in[11];

    const int M = BATCH * SEQ;                  // 8192
    const int N = D_MODEL;                      // 1024
    const int K = D_MODEL;                      // 1024

    const size_t tens = (size_t)M * D_MODEL;    // 8,388,608 elems
    const size_t wmat = (size_t)D_MODEL * D_MODEL;

    _Float16* q16  = (_Float16*)d_ws;           // f16 scratch layout
    _Float16* k16  = q16  + tens;
    _Float16* v16  = k16  + tens;
    _Float16* Wq16 = v16  + tens;
    _Float16* Wk16 = Wq16 + wmat;
    _Float16* Wv16 = Wk16 + wmat;
    _Float16* W016 = Wv16 + wmat;
    _Float16* Qp   = W016 + wmat;
    _Float16* Kpj  = Qp   + tens;
    _Float16* Vpj  = Kpj  + tens;
    _Float16* Ctx  = Vpj  + tens;               // ~126 MB total

    const dim3 tBlk(256);

    // f32 -> f16 conversions (inputs once, weights once — reused by 64 blocks)
    cvt_kernel<<<(int)(tens / 8 / 256), tBlk, 0, stream>>>(q,  q16, (int)(tens / 8));
    cvt_kernel<<<(int)(tens / 8 / 256), tBlk, 0, stream>>>(k,  k16, (int)(tens / 8));
    cvt_kernel<<<(int)(tens / 8 / 256), tBlk, 0, stream>>>(v,  v16, (int)(tens / 8));
    cvt_kernel<<<(int)(wmat / 8 / 256), tBlk, 0, stream>>>(Wq, Wq16, (int)(wmat / 8));
    cvt_kernel<<<(int)(wmat / 8 / 256), tBlk, 0, stream>>>(Wk, Wk16, (int)(wmat / 8));
    cvt_kernel<<<(int)(wmat / 8 / 256), tBlk, 0, stream>>>(Wv, Wv16, (int)(wmat / 8));
    cvt_kernel<<<(int)(wmat / 8 / 256), tBlk, 0, stream>>>(W0, W016, (int)(wmat / 8));

    const dim3 gGemm(M / BM, N / BN);           // 64 x 8

    // Q/K/V projections (f16 in, f16 head-layout out)
    gemm_kernel<true><<<gGemm, tBlk, 0, stream>>>(q16, Wq16, bq, Qp,  M, N, K);
    gemm_kernel<true><<<gGemm, tBlk, 0, stream>>>(k16, Wk16, bk, Kpj, M, N, K);
    gemm_kernel<true><<<gGemm, tBlk, 0, stream>>>(v16, Wv16, bv, Vpj, M, N, K);

    // Attention: (S/128, H, B) blocks
    attn_kernel<<<dim3(SEQ / 128, NHEAD, BATCH), tBlk, 0, stream>>>(Qp, Kpj, Vpj, Ctx);

    // Output projection (f16 in, f32 out)
    gemm_kernel<false><<<gGemm, tBlk, 0, stream>>>(Ctx, W016, b0, d_out, M, N, K);
}
